// AdaptiveChannelAttention_9663676416298
// MI455X (gfx1250) — matrix operand order repaired and checked by reading, not compiled, not run on hardware
//
#include <hip/hip_runtime.h>
#include <hip/hip_bf16.h>
#include <stdint.h>

// ---------------------------------------------------------------------------
// CDNA5 (gfx1250) implementation of AdaptiveChannelAttention forward.
// GEMMs on v_wmma_f32_16x16x32_bf16 (wave32); attn tile staged via TDM
// (tensor_load_to_lds) with hardware LDS padding.
// B=8, C=256, H=W=64, N=4096, heads=8, head_dim=32, 3C=768.
// ---------------------------------------------------------------------------

typedef __attribute__((ext_vector_type(16))) __bf16   bf16x16;
typedef __attribute__((ext_vector_type(8)))  float    f32x8;
typedef __attribute__((ext_vector_type(4)))  uint32_t u32x4;
typedef __attribute__((ext_vector_type(8)))  uint32_t u32x8;

union FragAB { bf16x16 v; uint32_t u[8]; __bf16 h[16]; };
union FragC  { f32x8   v; float    f[8]; };
union U16    { uint4   q; __bf16   h[8]; uint32_t u[4]; };
union BfPair { uint32_t u; __bf16  h[2]; };

// A-matrix 16x32 bf16 fragment (row-major row in LDS, K along the row):
// lanes 0-15 take K dwords {0..7}; lanes 16-31 take {8..15}.
__device__ __forceinline__ void frag_load_A(FragAB& fr, const __bf16* row, int lane) {
  const uint32_t* r = (const uint32_t*)row;
  const int o = (lane >= 16) ? 8 : 0;
#pragma unroll
  for (int i = 0; i < 4; ++i) { fr.u[i] = r[o + i]; fr.u[4 + i] = r[o + 4 + i]; }
}
// B-matrix 32x16 bf16 fragment (row = this lane's output column, K along row):
// lanes 0-15 take K dwords {0..7}; lanes 16-31 take {8..15}.
__device__ __forceinline__ void frag_load_B(FragAB& fr, const __bf16* row, int lane) {
  const uint32_t* r = (const uint32_t*)row;
  const int o = (lane >= 16) ? 8 : 0;
#pragma unroll
  for (int i = 0; i < 8; ++i) fr.u[i] = r[o + i];
}

__device__ __forceinline__ f32x8 wmma_bf16(const FragAB& a, const FragAB& b, const f32x8& c) {
  return __builtin_amdgcn_wmma_f32_16x16x32_bf16(false, a.v, false, b.v, (short)0, c, false, false);
}

__device__ __forceinline__ float gelu_f(float x)    { return 0.5f * x * (1.f + erff(x * 0.70710678118654752f)); }
__device__ __forceinline__ float sigmoid_f(float x) { return 1.f / (1.f + __expf(-x)); }

// ---------------------------------------------------------------------------
// init: zero the atomic accumulators (ssq[2][8][256], gap[8][256])
// ---------------------------------------------------------------------------
__global__ void k_init(float* __restrict__ ssq, float* __restrict__ gap) {
  const int i = blockIdx.x * 256 + threadIdx.x;
  if (i < 4096) ssq[i] = 0.f;
  if (i < 2048) gap[i] = 0.f;
}

// ---------------------------------------------------------------------------
// qkv = xt @ w_qkv   (per batch: 4096x256 @ 256x768 -> bf16 (B,N,768))
// macro tile 128(n) x 64(m); 8 waves in 4x2; each wave = 2x2 WMMA frags.
// A staged with packed bf16 pairs (ds_store_b32 + v_cvt_pk_bf16_f32).
// ---------------------------------------------------------------------------
__global__ __launch_bounds__(256) void k_gemm_qkv(const float* __restrict__ x,
                                                  const float* __restrict__ wqkv,
                                                  __bf16* __restrict__ qkv) {
  const int b  = blockIdx.z;
  const int n0 = blockIdx.x * 128;
  const int m0 = blockIdx.y * 64;
  __shared__ __bf16 As[128 * 40];   // rows n, K=c along row, pad 40 (20 dwords)
  __shared__ __bf16 Bs[64 * 40];    // rows m (B columns), K=c along row
  uint32_t* As32 = (uint32_t*)As;
  const int tid = threadIdx.x, lane = tid & 31, w = tid >> 5;
  const int wn = w & 3, wm = w >> 2;
  FragC acc[2][2];
#pragma unroll
  for (int i = 0; i < 2; ++i)
#pragma unroll
    for (int j = 0; j < 2; ++j)
#pragma unroll
      for (int r = 0; r < 8; ++r) acc[i][j].f[r] = 0.f;

  for (int k0 = 0; k0 < 256; k0 += 32) {
    // A tile: x[b][k0+c][n0+n] -> As[n][c]; two consecutive c per thread,
    // packed into one dword per n (4 ds_store_b32 per task).
#pragma unroll
    for (int j = 0; j < 2; ++j) {
      const int idx = tid + j * 256;          // 512 tasks: 16 c-pairs x 32 n4
      const int cp = idx >> 5;                // c-pair 0..15
      const int n4 = (idx & 31) << 2;         // 0..124
      const float4 x0 = *(const float4*)&x[((size_t)b * 256 + k0 + cp * 2    ) * 4096 + n0 + n4];
      const float4 x1 = *(const float4*)&x[((size_t)b * 256 + k0 + cp * 2 + 1) * 4096 + n0 + n4];
      BfPair p0, p1, p2, p3;
      p0.h[0] = (__bf16)x0.x; p0.h[1] = (__bf16)x1.x;
      p1.h[0] = (__bf16)x0.y; p1.h[1] = (__bf16)x1.y;
      p2.h[0] = (__bf16)x0.z; p2.h[1] = (__bf16)x1.z;
      p3.h[0] = (__bf16)x0.w; p3.h[1] = (__bf16)x1.w;
      As32[(n4 + 0) * 20 + cp] = p0.u;
      As32[(n4 + 1) * 20 + cp] = p1.u;
      As32[(n4 + 2) * 20 + cp] = p2.u;
      As32[(n4 + 3) * 20 + cp] = p3.u;
    }
    // B tile: wqkv[k0+c][m0+m] -> Bs[m][c]
#pragma unroll
    for (int j = 0; j < 8; ++j) {
      const int idx = tid + j * 256;          // 2048
      const int m = idx & 63, c = idx >> 6;
      Bs[m * 40 + c] = (__bf16)wqkv[(size_t)(k0 + c) * 768 + m0 + m];
    }
    __syncthreads();
    FragAB af[2], bf2[2];
#pragma unroll
    for (int i = 0; i < 2; ++i) frag_load_A(af[i],  &As[(wn * 32 + i * 16 + (lane & 15)) * 40], lane);
#pragma unroll
    for (int i = 0; i < 2; ++i) frag_load_B(bf2[i], &Bs[(wm * 32 + i * 16 + (lane & 15)) * 40], lane);
#pragma unroll
    for (int ni = 0; ni < 2; ++ni)
#pragma unroll
      for (int mi = 0; mi < 2; ++mi)
        acc[ni][mi].v = wmma_bf16(af[ni], bf2[mi], acc[ni][mi].v);
    __syncthreads();
  }
#pragma unroll
  for (int ni = 0; ni < 2; ++ni)
#pragma unroll
    for (int mi = 0; mi < 2; ++mi) {
      const int m  = m0 + wm * 32 + mi * 16 + (lane & 15);
      const int nb = n0 + wn * 32 + ni * 16 + ((lane >= 16) ? 8 : 0);
#pragma unroll
      for (int r = 0; r < 8; ++r)
        qkv[((size_t)b * 4096 + nb + r) * 768 + m] = (__bf16)acc[ni][mi].f[r];
    }
}

// ---------------------------------------------------------------------------
// ssq[qk][b][c] = sum_n qkv[b][n][qk*256 + c]^2   (for l2-normalize folding)
// ---------------------------------------------------------------------------
__global__ __launch_bounds__(256) void k_sumsq(const __bf16* __restrict__ qkv,
                                               float* __restrict__ ssq) {
  const int b = blockIdx.z, qk = blockIdx.y, nc = blockIdx.x;
  const int c = threadIdx.x;
  const __bf16* p = qkv + (size_t)b * 4096 * 768 + qk * 256 + c;
  float s = 0.f;
  for (int n = nc * 512; n < nc * 512 + 512; ++n) {
    const float v = (float)p[(size_t)n * 768];
    s += v * v;
  }
  atomicAdd(&ssq[qk * 2048 + b * 256 + c], s);
}

// ---------------------------------------------------------------------------
// attention logits S = q @ k^T (K=4096 split over 8 waves) fused with
// scale (rsq_q * rsq_k * temperature) and row softmax -> bf16 attn (B,h,32,32)
// ---------------------------------------------------------------------------
__global__ __launch_bounds__(256) void k_attn(const __bf16* __restrict__ qkv,
                                              const float* __restrict__ ssq,
                                              const float* __restrict__ temp,
                                              __bf16* __restrict__ attnw) {
  __shared__ uint4 smem4[2176];                 // 34816 B, re-used for reduction
  __bf16* tiles = (__bf16*)smem4;
  float*  red   = (float*)smem4;                // 8*1024 f32 (overlaps tiles, sync'd)
  const int bh = blockIdx.x;
  const int b = bh >> 3, h = bh & 7;
  const int tid = threadIdx.x, lane = tid & 31, w = tid >> 5;
  __bf16* tQ = tiles + w * 2176;                // per-wave private 32x34 tiles
  __bf16* tK = tQ + 1088;

  FragC acc[2][2];
#pragma unroll
  for (int i = 0; i < 2; ++i)
#pragma unroll
    for (int j = 0; j < 2; ++j)
#pragma unroll
      for (int r = 0; r < 8; ++r) acc[i][j].f[r] = 0.f;

  const __bf16* base = qkv + (size_t)b * 4096 * 768;
  const int qo = h * 32, ko = 256 + h * 32;
  for (int it = 0; it < 16; ++it) {
    const int n = w * 512 + it * 32 + lane;
    const __bf16* pr = base + (size_t)n * 768;
    U16 q0, q1, q2, q3, kk0, kk1, kk2, kk3;
    q0.q  = *(const uint4*)(pr + qo);      q1.q  = *(const uint4*)(pr + qo + 8);
    q2.q  = *(const uint4*)(pr + qo + 16); q3.q  = *(const uint4*)(pr + qo + 24);
    kk0.q = *(const uint4*)(pr + ko);      kk1.q = *(const uint4*)(pr + ko + 8);
    kk2.q = *(const uint4*)(pr + ko + 16); kk3.q = *(const uint4*)(pr + ko + 24);
#pragma unroll
    for (int c = 0; c < 8; ++c) {           // transpose into [c][n] tiles
      tQ[(c     ) * 34 + lane] = q0.h[c];
      tQ[(c +  8) * 34 + lane] = q1.h[c];
      tQ[(c + 16) * 34 + lane] = q2.h[c];
      tQ[(c + 24) * 34 + lane] = q3.h[c];
      tK[(c     ) * 34 + lane] = kk0.h[c];
      tK[(c +  8) * 34 + lane] = kk1.h[c];
      tK[(c + 16) * 34 + lane] = kk2.h[c];
      tK[(c + 24) * 34 + lane] = kk3.h[c];
    }
    FragAB aq[2], bk[2];
#pragma unroll
    for (int i = 0; i < 2; ++i) frag_load_A(aq[i], tQ + (i * 16 + (lane & 15)) * 34, lane);
#pragma unroll
    for (int i = 0; i < 2; ++i) frag_load_B(bk[i], tK + (i * 16 + (lane & 15)) * 34, lane);
#pragma unroll
    for (int cH = 0; cH < 2; ++cH)
#pragma unroll
      for (int dH = 0; dH < 2; ++dH)
        acc[cH][dH].v = wmma_bf16(aq[cH], bk[dH], acc[cH][dH].v);
  }
  __syncthreads();                              // everyone done with tiles
#pragma unroll
  for (int cH = 0; cH < 2; ++cH)
#pragma unroll
    for (int dH = 0; dH < 2; ++dH)
#pragma unroll
      for (int r = 0; r < 8; ++r) {
        const int c = cH * 16 + r + ((lane >= 16) ? 8 : 0);
        const int d = dH * 16 + (lane & 15);
        red[w * 1024 + c * 32 + d] = acc[cH][dH].f[r];
      }
  __syncthreads();
  for (int e = tid; e < 1024; e += 256) {
    float s = 0.f;
#pragma unroll
    for (int ww = 0; ww < 8; ++ww) s += red[ww * 1024 + e];
    red[e] = s;                                 // only this thread touches e
  }
  __syncthreads();
  const float tt = temp[h];
  const float sk = rsqrtf(fmaxf(ssq[2048 + b * 256 + h * 32 + lane], 1e-12f));
#pragma unroll
  for (int rr = 0; rr < 4; ++rr) {              // wave w owns rows 4w..4w+3
    const int c = w * 4 + rr;
    const float sq = rsqrtf(fmaxf(ssq[b * 256 + h * 32 + c], 1e-12f));
    float v = red[c * 32 + lane] * sq * sk * tt;
    float mx = v;
#pragma unroll
    for (int o = 16; o >= 1; o >>= 1) mx = fmaxf(mx, __shfl_xor(mx, o));
    const float ev = __expf(v - mx);
    float sum = ev;
#pragma unroll
    for (int o = 16; o >= 1; o >>= 1) sum += __shfl_xor(sum, o);
    attnw[((size_t)bh * 32 + c) * 32 + lane] = (__bf16)(ev / sum);
  }
}

// ---------------------------------------------------------------------------
// attened[b,n,h*32+c] = sum_d attn[b,h,c,d] * v[b,h,d,n]  (K=32 -> one WMMA)
// attn tile DMA'd into LDS by the Tensor Data Mover with hardware row
// padding (16 dwords data + 1 dword pad => 34-bf16 rows); B fragment
// streams straight from global (contiguous per lane).
// ---------------------------------------------------------------------------
__global__ __launch_bounds__(256) void k_att(const __bf16* __restrict__ qkv,
                                             const __bf16* __restrict__ attnw,
                                             __bf16* __restrict__ attened) {
  const int b = blockIdx.z, h = blockIdx.y;
  __shared__ __bf16 At[32 * 34];
  const int tid = threadIdx.x, lane = tid & 31, w = tid >> 5;
  const __bf16* asrc = attnw + (size_t)(b * 8 + h) * 1024;

  if (w == 0) {
    // Tensor DMA descriptor (D#), ISA ch.8: 32x32 bf16 tile, LDS padding
    // of 1 dword after every 16 dwords -> row stride 34 bf16.
    const uint64_t ga = (uint64_t)(uintptr_t)asrc;
    const uint32_t galo = (uint32_t)__builtin_amdgcn_readfirstlane((int)(uint32_t)ga);
    const uint32_t gahi = (uint32_t)__builtin_amdgcn_readfirstlane((int)(uint32_t)(ga >> 32));
    const uint32_t la   = (uint32_t)__builtin_amdgcn_readfirstlane((int)(uint32_t)(uintptr_t)At);
    u32x4 g0;
    g0.x = 1u;                                  // count=1 (valid user descriptor)
    g0.y = la;                                  // lds_addr (bytes)
    g0.z = galo;                                // global_addr[31:0]
    g0.w = (gahi & 0x01FFFFFFu) | (2u << 30);   // global_addr[56:32] | type=2
    u32x8 g1;
    g1.s0 = (1u << 16)                          // data_size = 2 bytes
          | (1u << 20)                          // pad_enable
          | (3u << 22)                          // pad_interval: 16 dwords
          | (0u << 25);                         // pad_amount: 1 dword
    g1.s1 = (32u & 0xFFFFu) << 16;              // tensor_dim0 = 32 (lo)
    g1.s2 = (32u << 16);                        // tensor_dim0 hi=0 | tensor_dim1 = 32
    g1.s3 = (32u << 16);                        // tensor_dim1 hi=0 | tile_dim0 = 32
    g1.s4 = 32u;                                // tile_dim1 = 32, tile_dim2 = 0
    g1.s5 = 32u;                                // tensor_dim0_stride = 32 (lo)
    g1.s6 = 0u;                                 // stride hi, tensor_dim1_stride lo
    g1.s7 = 0u;
    asm volatile("tensor_load_to_lds %0, %1" :: "s"(g0), "s"(g1) : "memory");
    __builtin_amdgcn_s_wait_tensorcnt(0);
  }
  __syncthreads();

  FragAB a0, a1, bf2;
  frag_load_A(a0, At + ((lane & 15)) * 34, lane);
  frag_load_A(a1, At + (16 + (lane & 15)) * 34, lane);
  const int nn  = blockIdx.x * 128 + w * 16 + (lane & 15);
  const int col = 512 + h * 32 + ((lane >= 16) ? 16 : 0);
  const uint4* p = (const uint4*)(qkv + ((size_t)b * 4096 + nn) * 768 + col);
  ((uint4*)bf2.u)[0] = p[0];
  ((uint4*)bf2.u)[1] = p[1];
  FragC acc0, acc1;
#pragma unroll
  for (int r = 0; r < 8; ++r) { acc0.f[r] = 0.f; acc1.f[r] = 0.f; }
  acc0.v = wmma_bf16(a0, bf2, acc0.v);
  acc1.v = wmma_bf16(a1, bf2, acc1.v);
  U16 o0, o1;
#pragma unroll
  for (int r = 0; r < 8; ++r) { o0.h[r] = (__bf16)acc0.f[r]; o1.h[r] = (__bf16)acc1.f[r]; }
  const int roff = (lane >= 16) ? 8 : 0;
  __bf16* dst = attened + ((size_t)b * 4096 + nn) * 256 + h * 32;
  *(uint4*)(dst + roff)      = o0.q;
  *(uint4*)(dst + 16 + roff) = o1.q;
}

// ---------------------------------------------------------------------------
// depthwise 3x3 conv on v + bias + BN + GELU -> convb bf16 (B,N,C)
// block: one (b, y, 32-channel chunk); LDS tile [3][66][32]
// ---------------------------------------------------------------------------
__global__ __launch_bounds__(256) void k_conv(const __bf16* __restrict__ qkv,
                                              const float* __restrict__ dwk,
                                              const float* __restrict__ dwb,
                                              const float* __restrict__ g,
                                              const float* __restrict__ be,
                                              const float* __restrict__ mu,
                                              const float* __restrict__ va,
                                              __bf16* __restrict__ convb) {
  const int c0 = blockIdx.x * 32, y = blockIdx.y, b = blockIdx.z;
  __shared__ __bf16 tile[3 * 66 * 32];
  const __bf16* base = qkv + (size_t)b * 4096 * 768 + 512;
  const int tid = threadIdx.x;
  for (int i = tid; i < 3 * 66 * 32; i += 256) {
    const int c = i & 31, xx = (i >> 5) % 66, ry = i / 2112;
    const int yy = y + ry - 1, xg = xx - 1;
    __bf16 v = (__bf16)0.f;
    if (yy >= 0 && yy < 64 && xg >= 0 && xg < 64)
      v = base[(size_t)(yy * 64 + xg) * 768 + c0 + c];
    tile[i] = v;
  }
  __syncthreads();
  const int c = tid & 31, xg = tid >> 5;        // 32 channels x 8 x-groups
  float wgt[9];
#pragma unroll
  for (int j = 0; j < 9; ++j) wgt[j] = dwk[(c0 + c) * 9 + j];
  const float bias = dwb[c0 + c];
  const float sc = rsqrtf(va[c0 + c] + 1e-3f) * g[c0 + c];
  const float mm = mu[c0 + c], bt = be[c0 + c];
#pragma unroll
  for (int i = 0; i < 8; ++i) {
    const int x = xg * 8 + i;
    float a = 0.f;
#pragma unroll
    for (int ry = 0; ry < 3; ++ry)
#pragma unroll
      for (int dx = 0; dx < 3; ++dx)
        a += wgt[ry * 3 + dx] * (float)tile[(ry * 66 + x + dx) * 32 + c];
    float v = (a + bias - mm) * sc + bt;
    v = gelu_f(v);
    convb[((size_t)b * 4096 + y * 64 + x) * 256 + c0 + c] = (__bf16)v;
  }
}

// ---------------------------------------------------------------------------
// gap[b][c] += sum over 256 n of attened (pre-gate)
// ---------------------------------------------------------------------------
__global__ __launch_bounds__(256) void k_gap(const __bf16* __restrict__ attened,
                                             float* __restrict__ gap) {
  const int b = blockIdx.y, c = threadIdx.x;
  const __bf16* p = attened + (size_t)b * 4096 * 256 + c;
  float s = 0.f;
  for (int n = blockIdx.x * 256; n < blockIdx.x * 256 + 256; ++n)
    s += (float)p[(size_t)n * 256];
  atomicAdd(&gap[b * 256 + c], s);
}

// ---------------------------------------------------------------------------
// channel interaction: sig_cm[b][c] = sigmoid(gelu(bn(gap/N @ w1 + b1)) @ w2 + b2)
// ---------------------------------------------------------------------------
__global__ __launch_bounds__(256) void k_cm(const float* __restrict__ gap,
                                            const float* __restrict__ w1, const float* __restrict__ b1,
                                            const float* __restrict__ g,  const float* __restrict__ be,
                                            const float* __restrict__ mu, const float* __restrict__ va,
                                            const float* __restrict__ w2, const float* __restrict__ b2,
                                            float* __restrict__ sig_cm) {
  __shared__ float gl[256];
  __shared__ float tmp[32];
  const int b = blockIdx.x, t = threadIdx.x;
  gl[t] = gap[b * 256 + t] * (1.f / 4096.f);
  __syncthreads();
  if (t < 32) {
    float s = b1[t];
    for (int c = 0; c < 256; ++c) s += gl[c] * w1[c * 32 + t];
    s = (s - mu[t]) * rsqrtf(va[t] + 1e-3f) * g[t] + be[t];
    tmp[t] = gelu_f(s);
  }
  __syncthreads();
  float s = b2[t];
#pragma unroll
  for (int o = 0; o < 32; ++o) s += tmp[o] * w2[o * 256 + t];
  sig_cm[b * 256 + t] = sigmoid_f(s);
}

// ---------------------------------------------------------------------------
// spatial interaction: sig_sm[b][n] from conv branch via two 1x1 convs
// ---------------------------------------------------------------------------
__global__ __launch_bounds__(256) void k_sm(const __bf16* __restrict__ convb,
                                            const float* __restrict__ w1, const float* __restrict__ b1,
                                            const float* __restrict__ g,  const float* __restrict__ be,
                                            const float* __restrict__ mu, const float* __restrict__ va,
                                            const float* __restrict__ w2, const float* __restrict__ b2,
                                            float* __restrict__ sig_sm) {
  __shared__ float sw1[4096];                   // si_w1 (256 x 16)
  const int tid = threadIdx.x;
  for (int i = tid; i < 4096; i += 256) sw1[i] = w1[i];
  __syncthreads();
  const int b = blockIdx.y;
  const int n = blockIdx.x * 256 + tid;
  const __bf16* row = convb + ((size_t)b * 4096 + n) * 256;
  float s1[16];
#pragma unroll
  for (int o = 0; o < 16; ++o) s1[o] = 0.f;
  for (int cc = 0; cc < 32; ++cc) {
    U16 u; u.q = *(const uint4*)(row + cc * 8);
#pragma unroll
    for (int j = 0; j < 8; ++j) {
      const float v = (float)u.h[j];
      const float* wr = &sw1[(cc * 8 + j) * 16];
#pragma unroll
      for (int o = 0; o < 16; ++o) s1[o] += v * wr[o];
    }
  }
  float sm = b2[0];
#pragma unroll
  for (int o = 0; o < 16; ++o) {
    float v = (s1[o] + b1[o] - mu[o]) * rsqrtf(va[o] + 1e-3f) * g[o] + be[o];
    sm += gelu_f(v) * w2[o];
  }
  sig_sm[b * 4096 + n] = sigmoid_f(sm);
}

// ---------------------------------------------------------------------------
// out = (attened*sig(sm) + conv*sig(cm)) @ w_proj + b_proj, stored as (B,C,N)
// gates fused into A-tile staging (bf16-pair dword loads + packed stores);
// bf16 WMMA; coalesced float4 stores.
// ---------------------------------------------------------------------------
__global__ __launch_bounds__(256) void k_gemm_proj(const __bf16* __restrict__ attened,
                                                   const __bf16* __restrict__ convb,
                                                   const float* __restrict__ sig_sm,
                                                   const float* __restrict__ sig_cm,
                                                   const float* __restrict__ wproj,
                                                   const float* __restrict__ bproj,
                                                   float* __restrict__ out) {
  const int b  = blockIdx.z;
  const int n0 = blockIdx.x * 128;
  const int m0 = blockIdx.y * 64;
  __shared__ __bf16 As[128 * 40];
  __shared__ __bf16 Bs[64 * 40];
  uint32_t* As32 = (uint32_t*)As;
  const int tid = threadIdx.x, lane = tid & 31, w = tid >> 5;
  const int wn = w & 3, wm = w >> 2;
  FragC acc[2][2];
#pragma unroll
  for (int i = 0; i < 2; ++i)
#pragma unroll
    for (int j = 0; j < 2; ++j)
#pragma unroll
      for (int r = 0; r < 8; ++r) acc[i][j].f[r] = 0.f;

  for (int k0 = 0; k0 < 256; k0 += 32) {
    // A tile: gate + add, bf16 pairs in and out (ds_store_b32)
#pragma unroll
    for (int j = 0; j < 8; ++j) {
      const int idx = tid + j * 256;            // 2048 tasks: 128 n x 16 c-pairs
      const int cp = idx & 15, nr = idx >> 4;
      const int n = n0 + nr;
      const size_t off = ((size_t)b * 4096 + n) * 256 + k0 + cp * 2;
      BfPair av, cv, res;
      av.u = *(const uint32_t*)&attened[off];
      cv.u = *(const uint32_t*)&convb[off];
      const float ss  = sig_sm[b * 4096 + n];
      const float cg0 = sig_cm[b * 256 + k0 + cp * 2];
      const float cg1 = sig_cm[b * 256 + k0 + cp * 2 + 1];
      res.h[0] = (__bf16)((float)av.h[0] * ss + (float)cv.h[0] * cg0);
      res.h[1] = (__bf16)((float)av.h[1] * ss + (float)cv.h[1] * cg1);
      As32[nr * 20 + cp] = res.u;
    }
#pragma unroll
    for (int j = 0; j < 8; ++j) {
      const int idx = tid + j * 256;
      const int m = idx & 63, cc = idx >> 6;
      Bs[m * 40 + cc] = (__bf16)wproj[(size_t)(k0 + cc) * 256 + m0 + m];
    }
    __syncthreads();
    FragAB af[2], bf2[2];
#pragma unroll
    for (int i = 0; i < 2; ++i) frag_load_A(af[i],  &As[(wn * 32 + i * 16 + (lane & 15)) * 40], lane);
#pragma unroll
    for (int i = 0; i < 2; ++i) frag_load_B(bf2[i], &Bs[(wm * 32 + i * 16 + (lane & 15)) * 40], lane);
#pragma unroll
    for (int ni = 0; ni < 2; ++ni)
#pragma unroll
      for (int mi = 0; mi < 2; ++mi)
        acc[ni][mi].v = wmma_bf16(af[ni], bf2[mi], acc[ni][mi].v);
    __syncthreads();
  }
#pragma unroll
  for (int ni = 0; ni < 2; ++ni)
#pragma unroll
    for (int mi = 0; mi < 2; ++mi) {
      const int m = m0 + wm * 32 + mi * 16 + (lane & 15);
      const float bias = bproj[m];
      const int nb = n0 + wn * 32 + ni * 16 + ((lane >= 16) ? 8 : 0);
      float* op = out + ((size_t)b * 256 + m) * 4096 + nb;
      const float4 lo = make_float4(acc[ni][mi].f[0] + bias, acc[ni][mi].f[1] + bias,
                                    acc[ni][mi].f[2] + bias, acc[ni][mi].f[3] + bias);
      const float4 hi = make_float4(acc[ni][mi].f[4] + bias, acc[ni][mi].f[5] + bias,
                                    acc[ni][mi].f[6] + bias, acc[ni][mi].f[7] + bias);
      *(float4*)op       = lo;
      *(float4*)(op + 4) = hi;
    }
}

// ---------------------------------------------------------------------------
// host launcher
// ---------------------------------------------------------------------------
extern "C" void kernel_launch(void* const* d_in, const int* in_sizes, int n_in,
                              void* d_out, int out_size, void* d_ws, size_t ws_size,
                              hipStream_t stream) {
  (void)in_sizes; (void)n_in; (void)out_size; (void)ws_size;
  const float* x        = (const float*)d_in[0];
  const float* w_qkv    = (const float*)d_in[1];
  const float* w_proj   = (const float*)d_in[2];
  const float* b_proj   = (const float*)d_in[3];
  const float* dw_k     = (const float*)d_in[4];
  const float* dw_b     = (const float*)d_in[5];
  const float* dw_gamma = (const float*)d_in[6];
  const float* dw_beta  = (const float*)d_in[7];
  const float* dw_mean  = (const float*)d_in[8];
  const float* dw_var   = (const float*)d_in[9];
  const float* ci_gamma = (const float*)d_in[10];
  const float* ci_beta  = (const float*)d_in[11];
  const float* ci_mean  = (const float*)d_in[12];
  const float* ci_var   = (const float*)d_in[13];
  const float* si_gamma = (const float*)d_in[14];
  const float* si_beta  = (const float*)d_in[15];
  const float* si_mean  = (const float*)d_in[16];
  const float* si_var   = (const float*)d_in[17];
  const float* ci_w1    = (const float*)d_in[18];
  const float* ci_b1    = (const float*)d_in[19];
  const float* ci_w2    = (const float*)d_in[20];
  const float* ci_b2    = (const float*)d_in[21];
  const float* si_w1    = (const float*)d_in[22];
  const float* si_b1    = (const float*)d_in[23];
  const float* si_w2    = (const float*)d_in[24];
  const float* si_b2    = (const float*)d_in[25];
  const float* temperature = (const float*)d_in[26];

  char* ws = (char*)d_ws;
  // workspace layout (bytes)
  __bf16* qkv     = (__bf16*)(ws);                       // 8*4096*768*2  = 50331648
  __bf16* attened = (__bf16*)(ws + 50331648);            // 8*4096*256*2  = 16777216
  __bf16* convb   = (__bf16*)(ws + 67108864);            // 16777216
  __bf16* attnw   = (__bf16*)(ws + 83886080);            // 8*8*32*32*2   = 131072
  float*  ssq     = (float*)(ws + 84017152);             // 2*8*256*4     = 16384
  float*  gap     = (float*)(ws + 84033536);             // 8*256*4       = 8192
  float*  sig_cm  = (float*)(ws + 84041728);             // 8192
  float*  sig_sm  = (float*)(ws + 84049920);             // 8*4096*4      = 131072

  k_init<<<24, 256, 0, stream>>>(ssq, gap);
  k_gemm_qkv<<<dim3(32, 12, 8), 256, 0, stream>>>(x, w_qkv, qkv);
  k_sumsq<<<dim3(8, 2, 8), 256, 0, stream>>>(qkv, ssq);
  k_attn<<<64, 256, 0, stream>>>(qkv, ssq, temperature, attnw);
  k_att<<<dim3(32, 8, 8), 256, 0, stream>>>(qkv, attnw, attened);
  k_conv<<<dim3(8, 64, 8), 256, 0, stream>>>(qkv, dw_k, dw_b, dw_gamma, dw_beta, dw_mean, dw_var, convb);
  k_gap<<<dim3(16, 8), 256, 0, stream>>>(attened, gap);
  k_cm<<<8, 256, 0, stream>>>(gap, ci_w1, ci_b1, ci_gamma, ci_beta, ci_mean, ci_var, ci_w2, ci_b2, sig_cm);
  k_sm<<<dim3(16, 8), 256, 0, stream>>>(convb, si_w1, si_b1, si_gamma, si_beta, si_mean, si_var, si_w2, si_b2, sig_sm);
  k_gemm_proj<<<dim3(32, 4, 8), 256, 0, stream>>>(attened, convb, sig_sm, sig_cm, w_proj, b_proj, (float*)d_out);
}
